// AudioVisualModel_2138893714283
// MI455X (gfx1250) — matrix operand order repaired and checked by reading, not compiled, not run on hardware
//
#include <hip/hip_runtime.h>
#include <hip/hip_bf16.h>

typedef _Float16 v16h __attribute__((ext_vector_type(16)));
typedef _Float16 v8h  __attribute__((ext_vector_type(8)));
typedef float    v8f  __attribute__((ext_vector_type(8)));
typedef int      v4i  __attribute__((vector_size(4 * sizeof(int))));

#define DIM 768
#define NA  512
#define NV  256
#define NB  32
// simgemm: 8 waves/WG, 16 M-rows per wave -> 128 rows/WG; NA/128 = 4 row-blocks/pair.

#ifndef __has_builtin
#define __has_builtin(x) 0
#endif

#if __has_builtin(__builtin_amdgcn_global_load_async_to_lds_b128)
#define HAVE_ASYNC_LDS 1
// imm offset applies to BOTH global and LDS addresses (ISA 15.18 async pseudocode)
#define ASYNC_CP16(g, l, off)                                                  \
  __builtin_amdgcn_global_load_async_to_lds_b128(                              \
      (__attribute__((address_space(1))) v4i*)(g),                             \
      (__attribute__((address_space(3))) v4i*)(l), (off), 0)
#else
#define HAVE_ASYNC_LDS 0
#endif

#if __has_builtin(__builtin_amdgcn_s_wait_asynccnt)
#define WAIT_ASYNC0() __builtin_amdgcn_s_wait_asynccnt(0)
#else
#define WAIT_ASYNC0() asm volatile("s_wait_asynccnt 0x0" ::: "memory")
#endif

__global__ __launch_bounds__(256) void normalize_f16_kernel(
    const float* __restrict__ in, _Float16* __restrict__ out, int nrows) {
  const int wave = threadIdx.x >> 5;
  const int lane = threadIdx.x & 31;
  const int row  = blockIdx.x * 8 + wave;
  if (row >= nrows) return;
  const float* x = in  + (size_t)row * DIM;
  _Float16*    y = out + (size_t)row * DIM;
  float v[24];
  float s = 0.f;
#pragma unroll
  for (int i = 0; i < 24; ++i) { v[i] = x[lane + 32 * i]; s += v[i] * v[i]; }
#pragma unroll
  for (int off = 16; off >= 1; off >>= 1) s += __shfl_xor(s, off, 32);
  const float inv = 1.0f / fmaxf(sqrtf(s), 1e-12f);
#pragma unroll
  for (int i = 0; i < 24; ++i) y[lane + 32 * i] = (_Float16)(v[i] * inv);
}

// Each WG: pair (b,c) and one 128-row block of audio tokens.
// Wave handles 16 audio rows x all 256 visual tokens: fused rowmax epilogue.
__global__ __launch_bounds__(256) void simgemm_kernel(
    const _Float16* __restrict__ A, const _Float16* __restrict__ V,
    float* __restrict__ partials) {
  // double-buffered V chunk: 256 cols x 32 K halfs, padded to 40 halfs (80B) per col
  __shared__ _Float16 lds[2][NV * 40];

  const int pair = blockIdx.y;          // 0..1023
  const int rb   = blockIdx.x;          // 0..3
  const int b = pair >> 5, c = pair & 31;
  const int wave = threadIdx.x >> 5, lane = threadIdx.x & 31;
  const int t = threadIdx.x;            // staging column id 0..255

  const _Float16* Abase = A + (size_t)(b * NA + rb * 128 + wave * 16) * DIM;
  const _Float16* Vbase = V + (size_t)c * NV * DIM;

  // A fragment addressing (16x32 f16): lanes 0-15 -> M=lane, K 0-15;
  // lanes 16-31 -> M=lane-16, K 16-31.
  const int arow  = lane & 15;
  const int akoff = (lane >> 4) * 16;
  const _Float16* arp = Abase + (size_t)arow * DIM + akoff;

  // B fragment addressing (32x16 f16 from LDS): lanes 0-15 -> N=lane, K 0-15;
  // lanes 16-31 -> N=lane-16, K 16-31.
  const int bcol  = lane & 15;
  const int bkoff = (lane >> 4) * 16;

  v8f acc[16];
  const v8f vz = {0.f, 0.f, 0.f, 0.f, 0.f, 0.f, 0.f, 0.f};
#pragma unroll
  for (int i = 0; i < 16; ++i) acc[i] = vz;

  const _Float16* gstage = Vbase + (size_t)t * DIM;  // this thread's V column
  _Float16* lstage[2] = {&lds[0][t * 40], &lds[1][t * 40]};

  // ---- stage K-chunk 0 ----
#if HAVE_ASYNC_LDS
  ASYNC_CP16(gstage, lstage[0], 0);
  ASYNC_CP16(gstage, lstage[0], 16);
  ASYNC_CP16(gstage, lstage[0], 32);
  ASYNC_CP16(gstage, lstage[0], 48);
  WAIT_ASYNC0();
#else
  {
    const v8h* g = (const v8h*)gstage;
    v8h r0 = g[0], r1 = g[1], r2 = g[2], r3 = g[3];
    v8h* s = (v8h*)lstage[0];
    s[0] = r0; s[1] = r1; s[2] = r2; s[3] = r3;
  }
#endif
  __syncthreads();

  for (int kc = 0; kc < 24; ++kc) {
    const int cur = kc & 1;
    const bool pf = (kc + 1) < 24;
#if HAVE_ASYNC_LDS
    if (pf) {  // async DMA next V chunk into the other buffer, overlapped with WMMAs
      const _Float16* gp = gstage + (kc + 1) * 32;
      _Float16* lp = lstage[1 - cur];
      ASYNC_CP16(gp, lp, 0);
      ASYNC_CP16(gp, lp, 16);
      ASYNC_CP16(gp, lp, 32);
      ASYNC_CP16(gp, lp, 48);
    }
#else
    v8h r0, r1, r2, r3;
    if (pf) {
      const v8h* g = (const v8h*)(gstage + (kc + 1) * 32);
      r0 = g[0]; r1 = g[1]; r2 = g[2]; r3 = g[3];
    }
#endif
    // A fragment for this K-chunk (two contiguous 16B loads)
    v16h afrag;
    {
      const _Float16* p = arp + kc * 32;
      v8h a0 = *(const v8h*)(p);
      v8h a1 = *(const v8h*)(p + 8);
#pragma unroll
      for (int i = 0; i < 8; ++i) { afrag[i] = a0[i]; afrag[i + 8] = a1[i]; }
    }
    const _Float16* lbase = &lds[cur][0];
#pragma unroll
    for (int n = 0; n < 16; ++n) {
      const _Float16* p = lbase + (n * 16 + bcol) * 40 + bkoff;
      v8h b0 = *(const v8h*)(p);
      v8h b1 = *(const v8h*)(p + 8);
      v16h bfrag;
#pragma unroll
      for (int i = 0; i < 8; ++i) { bfrag[i] = b0[i]; bfrag[i + 8] = b1[i]; }
      acc[n] = __builtin_amdgcn_wmma_f32_16x16x32_f16(
          false, afrag, false, bfrag, (short)0, acc[n], false, false);
    }
#if HAVE_ASYNC_LDS
    WAIT_ASYNC0();   // my async writes into buf[1-cur] are done
#else
    if (pf) {
      v8h* s = (v8h*)lstage[1 - cur];
      s[0] = r0; s[1] = r1; s[2] = r2; s[3] = r3;
    }
#endif
    __syncthreads();  // everyone's writes visible, everyone's reads of buf[cur] done
  }

  // Epilogue: per audio row, max over all 256 visual tokens, then row-sum.
  // C layout: acc[n][r] at lane L holds M = r + 8*(L>=16), N = n*16 + (L&15).
  float mx[8];
#pragma unroll
  for (int r = 0; r < 8; ++r) {
    float m = acc[0][r];
#pragma unroll
    for (int n = 1; n < 16; ++n) m = fmaxf(m, acc[n][r]);
    mx[r] = m;
  }
#pragma unroll
  for (int off = 1; off < 16; off <<= 1) {
#pragma unroll
    for (int r = 0; r < 8; ++r) mx[r] = fmaxf(mx[r], __shfl_xor(mx[r], off, 32));
  }
  float rs = 0.f;
#pragma unroll
  for (int r = 0; r < 8; ++r) rs += mx[r];
  // lane 0 holds rows 0-7 sum, lane 16 rows 8-15 sum (deterministic, no atomics)
  if ((lane & 15) == 0)
    partials[(size_t)pair * 64 + rb * 16 + wave * 2 + (lane >> 4)] = rs;
}

__global__ __launch_bounds__(256) void reduce_clip_kernel(
    const float* __restrict__ part, float* __restrict__ clip) {
  const int p = blockIdx.x * blockDim.x + threadIdx.x;
  if (p < NB * NB) {
    float s = 0.f;
#pragma unroll
    for (int i = 0; i < 64; ++i) s += part[(size_t)p * 64 + i];
    clip[p] = s * (1.0f / (float)NA);
  }
}

__global__ __launch_bounds__(32) void loss_kernel(
    const float* __restrict__ clip, float* __restrict__ out) {
  const int i = threadIdx.x & 31;  // lane i owns row i and column i
  float rm = -3.0e38f, cm = -3.0e38f;
  for (int j = 0; j < NB; ++j) {
    rm = fmaxf(rm, 10.0f * clip[i * NB + j]);
    cm = fmaxf(cm, 10.0f * clip[j * NB + i]);
  }
  float rsum = 0.f, csum = 0.f;
  for (int j = 0; j < NB; ++j) {
    rsum += expf(10.0f * clip[i * NB + j] - rm);
    csum += expf(10.0f * clip[j * NB + i] - cm);
  }
  const float lse_r = rm + logf(rsum);
  const float lse_c = cm + logf(csum);
  const float diag  = 10.0f * clip[i * NB + i];
  float li = -0.5f * ((diag - lse_r) + (diag - lse_c));
#pragma unroll
  for (int off = 16; off >= 1; off >>= 1) li += __shfl_xor(li, off, 32);
  if (i == 0) out[0] = li * (1.0f / (float)NB);
}

extern "C" void kernel_launch(void* const* d_in, const int* in_sizes, int n_in,
                              void* d_out, int out_size, void* d_ws, size_t ws_size,
                              hipStream_t stream) {
  const float* audio  = (const float*)d_in[0];  // (32,512,768) f32
  const float* visual = (const float*)d_in[1];  // (32,256,768) f32

  char* ws = (char*)d_ws;
  const size_t a_bytes = (size_t)NB * NA * DIM * sizeof(_Float16);  // 25.2 MB
  const size_t v_bytes = (size_t)NB * NV * DIM * sizeof(_Float16);  // 12.6 MB
  const size_t p_bytes = (size_t)NB * NB * 64 * sizeof(float);      // 256 KB
  _Float16* Af16 = (_Float16*)ws;
  _Float16* Vf16 = (_Float16*)(ws + a_bytes);
  float* partials = (float*)(ws + a_bytes + v_bytes);
  float* clip     = (float*)(ws + a_bytes + v_bytes + p_bytes);

  normalize_f16_kernel<<<NB * NA / 8, 256, 0, stream>>>(audio, Af16, NB * NA);
  normalize_f16_kernel<<<NB * NV / 8, 256, 0, stream>>>(visual, Vf16, NB * NV);

  dim3 grid(NA / 128, NB * NB);
  simgemm_kernel<<<grid, 256, 0, stream>>>(Af16, Vf16, partials);

  reduce_clip_kernel<<<(NB * NB + 255) / 256, 256, 0, stream>>>(partials, clip);
  loss_kernel<<<1, 32, 0, stream>>>(clip, (float*)d_out);
}
